// MambaBlock_86414741996190
// MI455X (gfx1250) — matrix operand order repaired and checked by reading, not compiled, not run on hardware
//
#include <hip/hip_runtime.h>
#include <hip/hip_bf16.h>

// ---- problem constants (match reference) ----
#define BATCH   2
#define SEQL    2048
#define DMODEL  768
#define DINNER  1536
#define DTRANK  16
#define DSTATE  16
#define DCONV   4
#define ROWS    (BATCH * SEQL)          // 4096 token rows

// ---- WMMA types (probe-confirmed signatures) ----
typedef __attribute__((ext_vector_type(16))) __bf16 v16bf;
typedef __attribute__((ext_vector_type(8)))  __bf16 v8bf;
typedef __attribute__((ext_vector_type(8)))  float  v8f;

// ---- GEMM tiling ----
#define MT   128   // rows per workgroup (8 waves x 16)
#define NT   64    // cols per workgroup (4 wmma subtiles per wave)
#define KT   32    // bf16 wmma K
#define LDK  40    // padded LDS row stride in halves (80B: bank-conflict free, 16B aligned)

__device__ __forceinline__ float silu_f(float x) {
    return x / (1.0f + __expf(-x));
}

// Generic tiled bf16 WMMA GEMM: C[M,N] = A[M,K] * Bw[K,N], f32 in/out.
// Double-buffered LDS panels: one barrier per K-step, global loads for panel
// k+1 overlap the WMMAs of panel k.
// mode 0: plain store to C0[M*N].
// mode 1: in_proj split: col < DINNER -> C0 (raw x_inner), else silu -> C1 (z).
__global__ __launch_bounds__(256)
void gemm_bf16_wmma(const float* __restrict__ A, const float* __restrict__ Bw,
                    float* __restrict__ C0, float* __restrict__ C1,
                    int M, int N, int K, int mode)
{
    __shared__ __align__(16) __bf16 sA[2][MT * LDK];   // row-major A panels
    __shared__ __align__(16) __bf16 sB[2][NT * LDK];   // transposed B panels: sB[n][k]

    const int tid  = threadIdx.x;
    const int lane = tid & 31;
    const int wave = tid >> 5;          // 0..7 -> M strip
    const int half = lane >> 4;         // 0 or 1 (wave32 half)
    const int l16  = lane & 15;
    const int m0   = blockIdx.y * MT;
    const int n0   = blockIdx.x * NT;

    // per-thread staging coordinates (constant across K-steps)
    const int ar  = tid >> 1;                 // A row 0..127 (2 threads/row)
    const int ac  = (tid & 1) * 16;           // A col group 0 or 16
    const int bk  = tid >> 4;                 // B k-row 0..15 (x2 groups below)
    const int bn4 = (tid & 15) * 4;           // B col group 0..60

    // ---- stage one K-panel into buffer `buf` ----
    auto stage = [&](int k0, int buf) {
        // A: 128x32 f32 -> bf16 (each thread: 16 contiguous floats = 4 float4)
        {
            const float* src = A + (size_t)(m0 + ar) * K + k0 + ac;
            __bf16* dst = &sA[buf][ar * LDK + ac];
            #pragma unroll
            for (int q = 0; q < 4; ++q) {
                const float4 v = *(const float4*)(src + q * 4);
                dst[q * 4 + 0] = (__bf16)v.x; dst[q * 4 + 1] = (__bf16)v.y;
                dst[q * 4 + 2] = (__bf16)v.z; dst[q * 4 + 3] = (__bf16)v.w;
            }
        }
        // B: 32x64 f32 -> bf16 transposed (coalesced in n, scatter to sB[n][k])
        #pragma unroll
        for (int g = 0; g < 2; ++g) {
            const int kr = bk + g * 16;       // 0..31
            float4 v = make_float4(0.f, 0.f, 0.f, 0.f);
            if (n0 + bn4 < N)                 // N always a multiple of 4
                v = *(const float4*)(Bw + (size_t)(k0 + kr) * N + n0 + bn4);
            __bf16* dst = &sB[buf][bn4 * LDK + kr];
            dst[0 * LDK] = (__bf16)v.x;
            dst[1 * LDK] = (__bf16)v.y;
            dst[2 * LDK] = (__bf16)v.z;
            dst[3 * LDK] = (__bf16)v.w;
        }
    };

    v8f acc[4] = {};                    // 4 n-subtiles of 16

    stage(0, 0);
    __syncthreads();

    const int ao_lo = half ? 8  : 0;    // ISA 16-bit A 16x32 layout offsets
    const int ao_hi = half ? 24 : 16;
    const int bo    = half ? 8 : 0;     // B k offset matching A: K{0..7,16..23} / K{8..15,24..31}

    for (int k0 = 0; k0 < K; k0 += KT) {
        const int cur = (k0 / KT) & 1;
        const int nxt = cur ^ 1;

        if (k0 + KT < K) {
            __builtin_prefetch(A + (size_t)(m0 + ar) * K + k0 + 2 * KT + ac, 0, 1);
            stage(k0 + KT, nxt);        // global loads overlap WMMAs below
        }

        // A fragment: lanes 0-15 take K{0..7,16..23}, lanes 16-31 K{8..15,24..31}
        const __bf16* pa = &sA[cur][(wave * 16 + l16) * LDK];
        v8bf alo = *(const v8bf*)(pa + ao_lo);
        v8bf ahi = *(const v8bf*)(pa + ao_hi);
        v16bf afrag = __builtin_shufflevector(alo, ahi,
            0, 1, 2, 3, 4, 5, 6, 7, 8, 9, 10, 11, 12, 13, 14, 15);

        // preload ALL B fragments, then issue WMMAs back-to-back
        v16bf bfrag[4];
        #pragma unroll
        for (int s = 0; s < 4; ++s) {
            const __bf16* pb = &sB[cur][(s * 16 + l16) * LDK + bo];
            v8bf blo = *(const v8bf*)(pb);
            v8bf bhi = *(const v8bf*)(pb + 16);
            bfrag[s] = __builtin_shufflevector(blo, bhi,
                0, 1, 2, 3, 4, 5, 6, 7, 8, 9, 10, 11, 12, 13, 14, 15);
        }
        #pragma unroll
        for (int s = 0; s < 4; ++s)
            acc[s] = __builtin_amdgcn_wmma_f32_16x16x32_bf16(
                false, afrag, false, bfrag[s], (short)0, acc[s], false, false);

        __syncthreads();                // releases cur for restaging, publishes nxt
    }

    // ---- store: C 16x16 f32 layout: lanes0-15 VGPRv = M=v, lanes16-31 M=8+v ----
    #pragma unroll
    for (int s = 0; s < 4; ++s) {
        int col = n0 + s * 16 + l16;
        if (col >= N) continue;
        #pragma unroll
        for (int v = 0; v < 8; ++v) {
            int row = m0 + wave * 16 + (half ? 8 : 0) + v;
            float val = acc[s][v];
            if (mode == 0) {
                C0[(size_t)row * N + col] = val;
            } else {                         // in_proj split + fused silu on z
                if (col < DINNER) C0[(size_t)row * DINNER + col] = val;
                else              C1[(size_t)row * DINNER + (col - DINNER)] = silu_f(val);
            }
        }
    }
}

// Causal depthwise conv1d (taps=4, left pad 3) + bias + silu.
__global__ __launch_bounds__(256)
void conv_silu_kernel(const float* __restrict__ xin, const float* __restrict__ cw,
                      const float* __restrict__ cb, float* __restrict__ xconv)
{
    const int d = blockIdx.x * 256 + threadIdx.x;   // 0..1535
    const int r = blockIdx.y;                       // 0..4095
    const int b = r >> 11;
    const int l = r & (SEQL - 1);
    float acc = cb[d];
    #pragma unroll
    for (int j = 0; j < DCONV; ++j) {
        int ll = l - (DCONV - 1) + j;
        if (ll >= 0)
            acc += cw[d * DCONV + j] * xin[((size_t)(b * SEQL + ll)) * DINNER + d];
    }
    xconv[(size_t)r * DINNER + d] = silu_f(acc);
}

// dt = softplus(dt16 @ W_dt + b_dt); K=16 so VALU with row broadcast via LDS.
__global__ __launch_bounds__(256)
void dt_proj_kernel(const float* __restrict__ xdbl, const float* __restrict__ Wdt,
                    const float* __restrict__ bdt, float* __restrict__ dtb)
{
    __shared__ float sdt[DTRANK];
    const int d = blockIdx.x * 256 + threadIdx.x;
    const int r = blockIdx.y;
    if (threadIdx.x < DTRANK) sdt[threadIdx.x] = xdbl[(size_t)r * 48 + threadIdx.x];
    __syncthreads();
    float acc = bdt[d];
    #pragma unroll
    for (int j = 0; j < DTRANK; ++j) acc += sdt[j] * Wdt[j * DINNER + d];
    dtb[(size_t)r * DINNER + d] = (acc > 20.f) ? acc : __logf(1.f + __expf(acc));
}

// Selective scan. One lane per (channel, state): lane = 16*sub + n, 2 channels/wave.
// Fuses the D-skip connection and the silu(z) gate (in place on zio).
__global__ __launch_bounds__(256)
void scan_kernel(const float* __restrict__ xconv, const float* __restrict__ dtb,
                 const float* __restrict__ xdbl, const float* __restrict__ Alog,
                 const float* __restrict__ Dp, float* __restrict__ zio)
{
    const int tid  = threadIdx.x;
    const int lane = tid & 31;
    const int n    = lane & 15;          // state index
    const int sub  = lane >> 4;          // which of 2 channels in this wave
    const int ch   = blockIdx.x * 16 + (tid >> 5) * 2 + sub;   // 0..3071
    const int b    = ch / DINNER;
    const int d    = ch % DINNER;

    const float An = -__expf(Alog[d * DSTATE + n]);
    const float Dd = Dp[d];
    const int rbase = b * SEQL;

    float h = 0.f;
    for (int t = 0; t < SEQL; ++t) {
        const int row = rbase + t;
        float u = 0.f, dtv = 0.f;
        if (n == 0) {                    // one VMEM read per channel, then broadcast
            u   = xconv[(size_t)row * DINNER + d];
            dtv = dtb[(size_t)row * DINNER + d];
        }
        u   = __shfl(u,   0, 16);
        dtv = __shfl(dtv, 0, 16);
        const float Bv = xdbl[(size_t)row * 48 + DTRANK + n];
        const float Cv = xdbl[(size_t)row * 48 + DTRANK + DSTATE + n];
        h = h * __expf(An * dtv) + (dtv * Bv) * u;
        float p = Cv * h;                // reduce over 16 state lanes
        p += __shfl_xor(p, 1, 16);
        p += __shfl_xor(p, 2, 16);
        p += __shfl_xor(p, 4, 16);
        p += __shfl_xor(p, 8, 16);
        if (n == 0) {
            const size_t idx = (size_t)row * DINNER + d;
            zio[idx] = (p + u * Dd) * zio[idx];   // y * silu(z), in place
        }
    }
}

extern "C" void kernel_launch(void* const* d_in, const int* in_sizes, int n_in,
                              void* d_out, int out_size, void* d_ws, size_t ws_size,
                              hipStream_t stream) {
    const float* x      = (const float*)d_in[0];   // [B,L,768]
    const float* W_in   = (const float*)d_in[1];   // [768,3072]
    const float* conv_w = (const float*)d_in[2];   // [1536,1,4]
    const float* conv_b = (const float*)d_in[3];   // [1536]
    const float* W_x    = (const float*)d_in[4];   // [1536,48]
    const float* W_dt   = (const float*)d_in[5];   // [16,1536]
    const float* b_dt   = (const float*)d_in[6];   // [1536]
    const float* A_log  = (const float*)d_in[7];   // [1536,16]
    const float* Dp     = (const float*)d_in[8];   // [1536]
    const float* W_out  = (const float*)d_in[9];   // [1536,768]
    float* out = (float*)d_out;                    // [B,L,768]

    // workspace partition (all f32)
    float* xin   = (float*)d_ws;                       // [4096,1536] raw x_inner
    float* zio   = xin   + (size_t)ROWS * DINNER;      // [4096,1536] silu(z) -> gated y
    float* xconv = zio   + (size_t)ROWS * DINNER;      // [4096,1536]
    float* dtb   = xconv + (size_t)ROWS * DINNER;      // [4096,1536]
    float* xdbl  = dtb   + (size_t)ROWS * DINNER;      // [4096,48]

    const dim3 blk(256);

    // 1) in_proj GEMM (split + fused silu on z): 4096x3072x768
    gemm_bf16_wmma<<<dim3(3072 / NT, ROWS / MT), blk, 0, stream>>>(
        x, W_in, xin, zio, ROWS, 2 * DINNER, DMODEL, 1);

    // 2) causal depthwise conv + silu
    conv_silu_kernel<<<dim3(DINNER / 256, ROWS), blk, 0, stream>>>(
        xin, conv_w, conv_b, xconv);

    // 3) x_dbl GEMM: 4096x48x1536 (N padded to one 64-tile)
    gemm_bf16_wmma<<<dim3(1, ROWS / MT), blk, 0, stream>>>(
        xconv, W_x, xdbl, nullptr, ROWS, DTRANK + 2 * DSTATE, DINNER, 0);

    // 4) dt projection + softplus
    dt_proj_kernel<<<dim3(DINNER / 256, ROWS), blk, 0, stream>>>(
        xdbl, W_dt, b_dt, dtb);

    // 5) selective scan + D skip + gate (in place on zio)
    scan_kernel<<<dim3(BATCH * DINNER / 16), blk, 0, stream>>>(
        xconv, dtb, xdbl, A_log, Dp, zio);

    // 6) out_proj GEMM: 4096x768x1536
    gemm_bf16_wmma<<<dim3(DMODEL / NT, ROWS / MT), blk, 0, stream>>>(
        zio, W_out, out, nullptr, ROWS, DMODEL, DINNER, 0);
}
